// EnhancedEGNNLayer_40458591928767
// MI455X (gfx1250) — matrix operand order repaired and checked by reading, not compiled, not run on hardware
//
#include <hip/hip_runtime.h>
#include <hip/hip_bf16.h>
#include <math.h>

// ---------------- problem constants ----------------
#define NN  20000
#define EE  320000
#define GG  16
#define DD  128
#define EDD 64
#define HH  256
#define RR  64

// d_out float offsets: x_new, pos_new, edge_attr_new, attn
#define OUT_X    0
#define OUT_POS  (NN*DD)                 // 2,560,000
#define OUT_EDGE (OUT_POS + NN*3)        // 2,620,000
#define OUT_ATTN (OUT_EDGE + (size_t)EE*EDD)

typedef __attribute__((ext_vector_type(16))) __bf16 bf16x16;
typedef __attribute__((ext_vector_type(8)))  __bf16 bf16x8;
typedef __attribute__((ext_vector_type(8)))  float  f32x8;

// ---------------- helpers ----------------
__device__ __forceinline__ unsigned short f2bf(float f) {
    unsigned u = __float_as_uint(f);
    unsigned r = u + 0x7FFFu + ((u >> 16) & 1u);   // RNE
    return (unsigned short)(r >> 16);
}
__device__ __forceinline__ float siluf(float v) { return v / (1.f + __expf(-v)); }

// ordered-uint encoding for float atomicMax (handles negatives)
__device__ __forceinline__ unsigned fenc(float f) {
    unsigned i = __float_as_uint(f);
    return (i & 0x80000000u) ? ~i : (i | 0x80000000u);
}
__device__ __forceinline__ float fdec(unsigned u) {
    return (u & 0x80000000u) ? __uint_as_float(u & 0x7fffffffu) : __uint_as_float(~u);
}
#define FENC_NEG_INF 0x007FFFFFu   // fenc(-inf)

__device__ __forceinline__ bf16x16 mk16(bf16x8 lo, bf16x8 hi) {
    bf16x16 r;
#pragma unroll
    for (int i = 0; i < 8; ++i) { r[i] = lo[i]; r[i + 8] = hi[i]; }
    return r;
}

// ---------------- weight prep: fp32 [K,Nout] -> bf16 WMMA-B-swizzled ----------------
// B fragment (32x16 bf16, K x N): lane l holds N = l&15, K-half = (l>>4)*16,
// vector element j <-> K = khalf + j.  Stored contiguously: 16 bf16 per lane.
__global__ void prep_w_kernel(const float* __restrict__ src, unsigned short* __restrict__ dst,
                              int K, int Nout) {
    int ntN = Nout >> 4;
    int kt = blockIdx.x / ntN;
    int nt = blockIdx.x % ntN;
    int lane = threadIdx.x;            // 0..31
    int n = nt * 16 + (lane & 15);
    int kbase = kt * 32 + ((lane >> 4) * 16);
    size_t o = ((size_t)(kt * ntN + nt) * 32 + lane) * 16;
#pragma unroll
    for (int j = 0; j < 16; ++j)
        dst[o + j] = f2bf(src[(size_t)(kbase + j) * Nout + n]);
}

// ---------------- generic WMMA GEMM (compile-time tiling) ----------------
// C[M,NOUT] = act(A[M,K] * W + bias) (+resid).  A bf16 row-major (lda), W pre-swizzled.
// Block = (NOUT/16/NT)*32 threads; every wave owns exactly NT 16-col tiles ->
// no divergent branches around WMMA, EXEC stays all-ones, K-loop fully unrolled.
template<int K, int NOUT, int NT>
__global__ void __launch_bounds__((NOUT/16/NT)*32)
gemm_bf16_t(const unsigned short* __restrict__ A, int lda,
            const unsigned short* __restrict__ Wswz,
            const float* __restrict__ bias, int act,
            const float* __restrict__ resid,
            float* __restrict__ outF,
            unsigned short* __restrict__ outB) {
    constexpr int NTN     = NOUT / 16;         // total 16-col tiles
    constexpr int NWAVE   = NTN / NT;          // waves per block
    constexpr int THREADS = NWAVE * 32;
    constexpr int KC8     = K / 8;             // 16B chunks per row
    constexpr int NK      = K / 32;            // K-steps

    __shared__ __align__(16) __bf16 sA[16 * K];

    const int row0 = blockIdx.x * 16;
    const int tid  = threadIdx.x;
    const int wave = tid >> 5;
    const int lane = tid & 31;

    // stage A panel [16 x K] into LDS (16B vector loads)
#pragma unroll
    for (int idx = tid; idx < 16 * KC8; idx += THREADS) {
        int r = idx / KC8, c = idx - r * KC8;
        ((int4*)sA)[idx] = *(const int4*)(A + (size_t)(row0 + r) * lda + c * 8);
    }
    __syncthreads();

    f32x8 acc[NT];
#pragma unroll
    for (int t = 0; t < NT; ++t)
        acc[t] = (f32x8){0.f,0.f,0.f,0.f,0.f,0.f,0.f,0.f};

    const int m     = lane & 15;
    const int khalf = (lane >> 4) * 16;
    const __bf16* parow = sA + m * K;
    // per-lane base into swizzled weights; advance by NTN*32*16 elems per K-step
    const unsigned short* pw = Wswz + (size_t)lane * 16;

#pragma unroll
    for (int kt = 0; kt < NK; ++kt) {
        bf16x16 afrag = mk16(*(const bf16x8*)(parow + kt * 32 + khalf),
                             *(const bf16x8*)(parow + kt * 32 + khalf + 8));
#pragma unroll
        for (int t = 0; t < NT; ++t) {
            const int nt = wave + t * NWAVE;
            const unsigned short* pb = pw + (size_t)(kt * NTN + nt) * 32 * 16;
            bf16x16 bfrag = mk16(*(const bf16x8*)pb, *(const bf16x8*)(pb + 8));
            acc[t] = __builtin_amdgcn_wmma_f32_16x16x32_bf16(false, afrag, false, bfrag,
                                                             (short)0, acc[t], false, false);
        }
    }

    // epilogue: D element r is row (lane>>4)*8 + r, col = lane&15 of tile
    const int mbase = row0 + ((lane >> 4) * 8);
#pragma unroll
    for (int t = 0; t < NT; ++t) {
        const int nt = wave + t * NWAVE;
        const int n  = nt * 16 + (lane & 15);
        const float bv = bias[n];
#pragma unroll
        for (int r = 0; r < 8; ++r) {
            float v = acc[t][r] + bv;
            if (act) v = siluf(v);
            size_t o = (size_t)(mbase + r) * NOUT + n;
            if (resid) v += resid[o];
            if (outF) outF[o] = v;
            if (outB) outB[o] = f2bf(v);
        }
    }
}

// ---------------- LayerNorm (block = 256 threads per row) ----------------
__global__ void ln_kernel(const float* __restrict__ in, int W,
                          const float* __restrict__ g, const float* __restrict__ b,
                          unsigned short* __restrict__ outB, float* __restrict__ outF) {
    __shared__ float s1[256], s2[256];
    const size_t row = blockIdx.x;
    const int tid = threadIdx.x;
    float sum = 0.f, sq = 0.f;
    for (int c = tid; c < W; c += 256) {
        float v = in[row * W + c];
        sum += v; sq += v * v;
    }
    s1[tid] = sum; s2[tid] = sq;
    __syncthreads();
    for (int s = 128; s > 0; s >>= 1) {
        if (tid < s) { s1[tid] += s1[tid + s]; s2[tid] += s2[tid + s]; }
        __syncthreads();
    }
    float mean = s1[0] / W;
    float var  = s2[0] / W - mean * mean;
    float rstd = rsqrtf(var + 1e-5f);
    for (int c = tid; c < W; c += 256) {
        float v = (in[row * W + c] - mean) * rstd * g[c] + b[c];
        if (outB) outB[row * W + c] = f2bf(v);
        if (outF) outF[row * W + c] = v;
    }
}

// ---------------- edge input assembly ----------------
__global__ void assemble_edge_kernel(const float* __restrict__ x, const float* __restrict__ pos,
                                     const float* __restrict__ ea, const float* __restrict__ u,
                                     const int* __restrict__ edge_index, const int* __restrict__ batch,
                                     const float* __restrict__ eu_means, const float* __restrict__ eu_betas,
                                     const float* __restrict__ rbf_means, const float* __restrict__ rbf_betas,
                                     unsigned short* __restrict__ eu_in, unsigned short* __restrict__ msg_in,
                                     unsigned short* __restrict__ rbf2b, float* __restrict__ xdiff) {
    const int e = blockIdx.x;
    const int row = edge_index[e];
    const int col = edge_index[EE + e];
    __shared__ float sd[3];
    if (threadIdx.x < 3) {
        float d = pos[row * 3 + threadIdx.x] - pos[col * 3 + threadIdx.x];
        sd[threadIdx.x] = d;
        xdiff[(size_t)e * 3 + threadIdx.x] = d;
    }
    __syncthreads();
    const float dist = sqrtf(sd[0]*sd[0] + sd[1]*sd[1] + sd[2]*sd[2]);
    const float cut  = (dist < 10.f) ? 0.5f * (__cosf(dist * 0.31415926535f) + 1.f) : 0.f;
    const float ex   = __expf(-0.5f * dist);   // alpha = 5/cutoff = 0.5
    const int ub = batch[row];

    for (int c = threadIdx.x; c < 640; c += 128) {
        float v;
        if (c < 128)       v = x[(size_t)row * DD + c];
        else if (c < 256)  v = x[(size_t)col * DD + (c - 128)];
        else if (c < 320)  { int r = c - 256; float d = ex - eu_means[r];
                             v = cut * __expf(-eu_betas[r] * d * d); }
        else if (c < 384)  v = ea[(size_t)e * EDD + (c - 320)];
        else               v = u[(size_t)ub * HH + (c - 384)];
        unsigned short bv = f2bf(v);
        eu_in[(size_t)e * 640 + c] = bv;
        if (c < 256) msg_in[(size_t)e * 384 + c] = bv;   // shared x_row|x_col prefix
    }
    for (int r = threadIdx.x; r < RR; r += 128) {
        float d = ex - rbf_means[r];
        unsigned short bv = f2bf(cut * __expf(-rbf_betas[r] * d * d));
        rbf2b[(size_t)e * RR + r] = bv;
        msg_in[(size_t)e * 384 + 320 + r] = bv;          // rbf2 block of msg_in
    }
}

// edge_attr_new (f32, in d_out) -> msg_in cols [256,320)
__global__ void msg_fill_kernel(const float* __restrict__ ean, unsigned short* __restrict__ msg_in) {
    size_t e = blockIdx.x; int t = threadIdx.x;
    msg_in[e * 384 + 256 + t] = f2bf(ean[e * EDD + t]);
}
// rbf2 -> msg_in cols [256,320) (turn msg_in prefix into coord_in, K=320, lda=384)
__global__ void coordify_kernel(const unsigned short* __restrict__ rbf2b, unsigned short* __restrict__ msg_in) {
    size_t e = blockIdx.x; int t = threadIdx.x;
    msg_in[e * 384 + 256 + t] = rbf2b[e * RR + t];
}

// ---------------- aggregation state init ----------------
__global__ void init_kernel(float* __restrict__ x_msg, float* __restrict__ denom,
                            unsigned* __restrict__ segmax, float* __restrict__ pos_delta) {
    size_t i = (size_t)blockIdx.x * blockDim.x + threadIdx.x;
    if (i < (size_t)NN * HH) x_msg[i] = 0.f;
    if (i < NN) { denom[i] = 0.f; segmax[i] = FENC_NEG_INF; }
    if (i < (size_t)NN * 3) pos_delta[i] = 0.f;
}

// ---------------- attention ----------------
__global__ void logits_kernel(const float* __restrict__ msgs, const float* __restrict__ attn_w,
                              const float* __restrict__ attn_b, const float* __restrict__ motif_imp,
                              const float* __restrict__ cross_bias,
                              const int* __restrict__ edge_index, const int* __restrict__ motif_types,
                              float* __restrict__ logits, unsigned* __restrict__ segmax) {
    const int e = blockIdx.x * 8 + (threadIdx.x >> 5);
    const int lane = threadIdx.x & 31;
    float s = 0.f;
#pragma unroll
    for (int i = 0; i < 8; ++i)
        s += msgs[(size_t)e * HH + lane + 32 * i] * attn_w[lane + 32 * i];
#pragma unroll
    for (int m = 16; m; m >>= 1) s += __shfl_xor(s, m, 32);
    if (lane == 0) {
        int row = edge_index[e], col = edge_index[EE + e];
        int ti = motif_types[row], tj = motif_types[col];
        float l = s + attn_b[0] + motif_imp[ti] + motif_imp[tj] + cross_bias[ti * 5 + tj];
        logits[e] = l;
        atomicMax(&segmax[col], fenc(l));
    }
}

__global__ void expv_kernel(float* __restrict__ logits, const unsigned* __restrict__ segmax,
                            float* __restrict__ denom, const int* __restrict__ edge_index) {
    size_t e = (size_t)blockIdx.x * blockDim.x + threadIdx.x;
    if (e >= EE) return;
    int col = edge_index[EE + e];
    float ev = __expf(logits[e] - fdec(segmax[col]));
    logits[e] = ev;
    atomicAdd(&denom[col], ev);
}

__global__ void attn_kernel(const float* __restrict__ expv, const float* __restrict__ denom,
                            const int* __restrict__ edge_index, float* __restrict__ attn_out) {
    size_t e = (size_t)blockIdx.x * blockDim.x + threadIdx.x;
    if (e >= EE) return;
    int col = edge_index[EE + e];
    attn_out[e] = expv[e] / (denom[col] + 1e-16f);
}

__global__ void scatter_msg_kernel(const float* __restrict__ msgs, const float* __restrict__ attn,
                                   const int* __restrict__ edge_index, float* __restrict__ x_msg) {
    size_t e = blockIdx.x; int t = threadIdx.x;
    int col = edge_index[EE + e];
    atomicAdd(&x_msg[(size_t)col * HH + t], msgs[e * HH + t] * attn[e]);
}

// ---------------- node path ----------------
__global__ void node_assemble_kernel(const float* __restrict__ x, const float* __restrict__ x_msg,
                                     unsigned short* __restrict__ node_in) {
    size_t n = blockIdx.x;
    for (int c = threadIdx.x; c < 384; c += 128) {
        float v = (c < 128) ? x[n * DD + c] : x_msg[n * HH + (c - 128)];
        node_in[n * 384 + c] = f2bf(v);
    }
}

// ---------------- coordinate path ----------------
__global__ void coord_dot_kernel(const float* __restrict__ h, const float* __restrict__ w2,
                                 const float* __restrict__ b2, const float* __restrict__ xdiff,
                                 const int* __restrict__ edge_index, float* __restrict__ pos_delta) {
    const int e = blockIdx.x * 8 + (threadIdx.x >> 5);
    const int lane = threadIdx.x & 31;
    float s = 0.f;
#pragma unroll
    for (int i = 0; i < 8; ++i)
        s += h[(size_t)e * HH + lane + 32 * i] * w2[lane + 32 * i];
#pragma unroll
    for (int m = 16; m; m >>= 1) s += __shfl_xor(s, m, 32);
    float cw = tanhf(s + b2[0]);
    if (lane < 3) {
        int row = edge_index[e];
        atomicAdd(&pos_delta[(size_t)row * 3 + lane], cw * xdiff[(size_t)e * 3 + lane]);
    }
}

__global__ void pos_update_kernel(const float* __restrict__ pos, const float* __restrict__ pos_delta,
                                  float* __restrict__ out) {
    size_t i = (size_t)blockIdx.x * blockDim.x + threadIdx.x;
    if (i < (size_t)NN * 3) out[i] = pos[i] + 0.1f * pos_delta[i];
}

// ---------------- host ----------------
extern "C" void kernel_launch(void* const* d_in, const int* in_sizes, int n_in,
                              void* d_out, int out_size, void* d_ws, size_t ws_size,
                              hipStream_t stream) {
    const float* x        = (const float*)d_in[0];
    const float* pos      = (const float*)d_in[1];
    const float* ea       = (const float*)d_in[2];
    const float* u        = (const float*)d_in[3];
    const float* eu_means = (const float*)d_in[4];
    const float* eu_betas = (const float*)d_in[5];
    const float* eu_w1    = (const float*)d_in[6];
    const float* eu_b1    = (const float*)d_in[7];
    const float* eu_ln_g  = (const float*)d_in[8];
    const float* eu_ln_b  = (const float*)d_in[9];
    const float* eu_w2    = (const float*)d_in[10];
    const float* eu_b2    = (const float*)d_in[11];
    const float* eu_w3    = (const float*)d_in[12];
    const float* eu_b3    = (const float*)d_in[13];
    const float* rbf_means= (const float*)d_in[14];
    const float* rbf_betas= (const float*)d_in[15];
    const float* msg_w1   = (const float*)d_in[16];
    const float* msg_b1   = (const float*)d_in[17];
    const float* msg_ln_g = (const float*)d_in[18];
    const float* msg_ln_b = (const float*)d_in[19];
    const float* msg_w2   = (const float*)d_in[20];
    const float* msg_b2   = (const float*)d_in[21];
    const float* attn_w   = (const float*)d_in[22];
    const float* attn_b   = (const float*)d_in[23];
    const float* motif_imp= (const float*)d_in[24];
    const float* cross_b  = (const float*)d_in[25];
    const float* node_w1  = (const float*)d_in[26];
    const float* node_b1  = (const float*)d_in[27];
    const float* node_ln_g= (const float*)d_in[28];
    const float* node_ln_b= (const float*)d_in[29];
    const float* node_w2  = (const float*)d_in[30];
    const float* node_b2  = (const float*)d_in[31];
    const float* coord_w1 = (const float*)d_in[32];
    const float* coord_b1 = (const float*)d_in[33];
    const float* coord_w2 = (const float*)d_in[34];
    const float* coord_b2 = (const float*)d_in[35];
    const float* norm_g   = (const float*)d_in[36];
    const float* norm_b   = (const float*)d_in[37];
    const int* edge_index = (const int*)d_in[38];
    const int* batch      = (const int*)d_in[39];
    const int* motif_t    = (const int*)d_in[40];

    float* out = (float*)d_out;
    float* out_edge = out + OUT_EDGE;
    float* out_attn = out + OUT_ATTN;

    // workspace carve-up (256B aligned)
    size_t off = 0;
    char* base = (char*)d_ws;
    auto alloc = [&](size_t bytes) -> char* {
        char* p = base + off;
        off += (bytes + 255) & ~(size_t)255;
        return p;
    };
    unsigned short* Weu1   = (unsigned short*)alloc((size_t)640*HH*2);
    unsigned short* Weu2   = (unsigned short*)alloc((size_t)HH*HH*2);
    unsigned short* Weu3   = (unsigned short*)alloc((size_t)HH*EDD*2);
    unsigned short* Wmsg1  = (unsigned short*)alloc((size_t)384*HH*2);
    unsigned short* Wmsg2  = (unsigned short*)alloc((size_t)HH*HH*2);
    unsigned short* Wnode1 = (unsigned short*)alloc((size_t)384*HH*2);
    unsigned short* Wnode2 = (unsigned short*)alloc((size_t)HH*DD*2);
    unsigned short* Wcoord1= (unsigned short*)alloc((size_t)320*HH*2);
    unsigned short* eu_in  = (unsigned short*)alloc((size_t)EE*640*2);
    unsigned short* msg_in = (unsigned short*)alloc((size_t)EE*384*2);
    unsigned short* rbf2b  = (unsigned short*)alloc((size_t)EE*RR*2);
    float*          xdiff  = (float*)         alloc((size_t)EE*3*4);
    float*          bufF   = (float*)         alloc((size_t)EE*HH*4);
    unsigned short* b16a   = (unsigned short*)alloc((size_t)EE*HH*2);
    unsigned short* b16b   = (unsigned short*)alloc((size_t)EE*HH*2);
    float*          logits = (float*)         alloc((size_t)EE*4);
    unsigned*       segmax = (unsigned*)      alloc((size_t)NN*4);
    float*          denom  = (float*)         alloc((size_t)NN*4);
    float*          x_msg  = (float*)         alloc((size_t)NN*HH*4);
    unsigned short* node_in= (unsigned short*)alloc((size_t)NN*384*2);
    float*          nodeF  = (float*)         alloc((size_t)NN*HH*4);
    unsigned short* node16 = (unsigned short*)alloc((size_t)NN*HH*2);
    float*          node2o = (float*)         alloc((size_t)NN*DD*4);
    float*       pos_delta = (float*)         alloc((size_t)NN*3*4);

    // 1. weight prep (bf16 + WMMA-B swizzle)
    prep_w_kernel<<<dim3((640/32)*(HH/16)), 32, 0, stream>>>(eu_w1,   Weu1,   640, HH);
    prep_w_kernel<<<dim3((HH/32)*(HH/16)),  32, 0, stream>>>(eu_w2,   Weu2,   HH,  HH);
    prep_w_kernel<<<dim3((HH/32)*(EDD/16)), 32, 0, stream>>>(eu_w3,   Weu3,   HH,  EDD);
    prep_w_kernel<<<dim3((384/32)*(HH/16)), 32, 0, stream>>>(msg_w1,  Wmsg1,  384, HH);
    prep_w_kernel<<<dim3((HH/32)*(HH/16)),  32, 0, stream>>>(msg_w2,  Wmsg2,  HH,  HH);
    prep_w_kernel<<<dim3((384/32)*(HH/16)), 32, 0, stream>>>(node_w1, Wnode1, 384, HH);
    prep_w_kernel<<<dim3((HH/32)*(DD/16)),  32, 0, stream>>>(node_w2, Wnode2, HH,  DD);
    prep_w_kernel<<<dim3((320/32)*(HH/16)), 32, 0, stream>>>(coord_w1,Wcoord1,320, HH);

    // 2. aggregation-state init
    init_kernel<<<dim3((NN*HH + 255)/256), 256, 0, stream>>>(x_msg, denom, segmax, pos_delta);

    // 3. edge input assembly (dist, RBFs, gathers -> bf16 panels)
    assemble_edge_kernel<<<dim3(EE), 128, 0, stream>>>(x, pos, ea, u, edge_index, batch,
        eu_means, eu_betas, rbf_means, rbf_betas, eu_in, msg_in, rbf2b, xdiff);

    // 4. edge update MLP
    gemm_bf16_t<640, 256, 2><<<dim3(EE/16), 256, 0, stream>>>(eu_in, 640, Weu1, eu_b1,
        1, nullptr, bufF, nullptr);
    ln_kernel<<<dim3(EE), 256, 0, stream>>>(bufF, HH, eu_ln_g, eu_ln_b, b16a, nullptr);
    gemm_bf16_t<256, 256, 2><<<dim3(EE/16), 256, 0, stream>>>(b16a, 256, Weu2, eu_b2,
        1, nullptr, nullptr, b16b);
    gemm_bf16_t<256, 64, 1><<<dim3(EE/16), 128, 0, stream>>>(b16b, 256, Weu3, eu_b3,
        0, ea, out_edge, nullptr);

    // 5. message MLP
    msg_fill_kernel<<<dim3(EE), 64, 0, stream>>>(out_edge, msg_in);
    gemm_bf16_t<384, 256, 2><<<dim3(EE/16), 256, 0, stream>>>(msg_in, 384, Wmsg1, msg_b1,
        1, nullptr, bufF, nullptr);
    ln_kernel<<<dim3(EE), 256, 0, stream>>>(bufF, HH, msg_ln_g, msg_ln_b, b16a, nullptr);
    gemm_bf16_t<256, 256, 2><<<dim3(EE/16), 256, 0, stream>>>(b16a, 256, Wmsg2, msg_b2,
        0, nullptr, bufF, nullptr);                     // bufF = messages

    // 6. segment softmax + message aggregation
    logits_kernel<<<dim3(EE/8), 256, 0, stream>>>(bufF, attn_w, attn_b, motif_imp, cross_b,
        edge_index, motif_t, logits, segmax);
    expv_kernel<<<dim3((EE + 255)/256), 256, 0, stream>>>(logits, segmax, denom, edge_index);
    attn_kernel<<<dim3((EE + 255)/256), 256, 0, stream>>>(logits, denom, edge_index, out_attn);
    scatter_msg_kernel<<<dim3(EE), HH, 0, stream>>>(bufF, out_attn, edge_index, x_msg);

    // 7. node MLP + residual + final norm
    node_assemble_kernel<<<dim3(NN), 128, 0, stream>>>(x, x_msg, node_in);
    gemm_bf16_t<384, 256, 2><<<dim3(NN/16), 256, 0, stream>>>(node_in, 384, Wnode1, node_b1,
        1, nullptr, nodeF, nullptr);
    ln_kernel<<<dim3(NN), 256, 0, stream>>>(nodeF, HH, node_ln_g, node_ln_b, node16, nullptr);
    gemm_bf16_t<256, 128, 1><<<dim3(NN/16), 256, 0, stream>>>(node16, 256, Wnode2, node_b2,
        0, x, node2o, nullptr);
    ln_kernel<<<dim3(NN), 256, 0, stream>>>(node2o, DD, norm_g, norm_b, nullptr, out + OUT_X);

    // 8. coordinate update (reuse msg_in prefix as coord_in, K=320 of lda=384)
    coordify_kernel<<<dim3(EE), 64, 0, stream>>>(rbf2b, msg_in);
    gemm_bf16_t<320, 256, 2><<<dim3(EE/16), 256, 0, stream>>>(msg_in, 384, Wcoord1, coord_b1,
        1, nullptr, bufF, nullptr);
    coord_dot_kernel<<<dim3(EE/8), 256, 0, stream>>>(bufF, coord_w2, coord_b2, xdiff,
        edge_index, pos_delta);
    pos_update_kernel<<<dim3((NN*3 + 255)/256), 256, 0, stream>>>(pos, pos_delta, out + OUT_POS);

    (void)in_sizes; (void)n_in; (void)out_size; (void)ws_size;
}
